// Head_53669911331118
// MI455X (gfx1250) — matrix operand order repaired and checked by reading, not compiled, not run on hardware
//
#include <hip/hip_runtime.h>
#include <cmath>

typedef _Float16 half_t;
typedef __attribute__((ext_vector_type(16))) _Float16 v16h;
typedef __attribute__((ext_vector_type(8)))  _Float16 v8h;
typedef __attribute__((ext_vector_type(4)))  _Float16 v4h;
typedef __attribute__((ext_vector_type(8)))  float    v8f;

constexpr int BB = 4;      // batch
constexpr int TT = 4096;   // sequence
constexpr int DD = 1024;   // embed
constexpr int HS = 64;     // head size
// softmax scale (1/sqrt(64)) * log2(e): softmax done in log2 domain so the
// hot loop uses raw v_exp_f32 (exp2) with no extra multiplies.
constexpr float QSCALE = 0.125f * 1.4426950408889634f;

// ---------------------------------------------------------------------------
// fp32 -> f16 conversion, 4 elements per thread (all sizes are multiples of 4)
// ---------------------------------------------------------------------------
__global__ void cvt_f32_to_f16(const float* __restrict__ src,
                               half_t* __restrict__ dst, int n) {
  int i = (blockIdx.x * blockDim.x + threadIdx.x) * 4;
  if (i >= n) return;
  float4 f = *(const float4*)(src + i);
  v4h h = { (half_t)f.x, (half_t)f.y, (half_t)f.z, (half_t)f.w };
  *(v4h*)(dst + i) = h;
}

// ---------------------------------------------------------------------------
// C[M,N] = oscale * Xh[M,D] * Wh[N,D]^T   (M = B*T, one 16x16 tile per wave)
// TRANS=0: store f16 row-major [M,N]        (used for Q, K)
// TRANS=1: store f16 transposed [B][N][T]   (used for V -> Vt)
// ---------------------------------------------------------------------------
template <int TRANS>
__global__ __launch_bounds__(128) void proj_gemm(
    const half_t* __restrict__ Xh, const half_t* __restrict__ Wh,
    half_t* __restrict__ Out, int N, float oscale) {
  const int lane   = threadIdx.x & 31;
  const int wave   = threadIdx.x >> 5;
  const int mt     = blockIdx.x;                 // 16-row tile of M
  const int nt     = blockIdx.y * 4 + wave;      // 16-col tile of N
  if (nt * 16 >= N) return;

  const int l16  = lane & 15;
  const int a_c0 = (lane >= 16) ? 16 : 0;        // A-frag K half base
  const int b_k0 = (lane >= 16) ? 16 : 0;        // B-frag K half base
  const int hb   = (lane >= 16) ? 8 : 0;         // C-frag row half base

  const half_t* arow = Xh + (size_t)(mt * 16 + l16) * DD;  // A row (token)
  const half_t* brow = Wh + (size_t)(nt * 16 + l16) * DD;  // B col (= W row)

  v8f acc = {};
#pragma unroll 4
  for (int d = 0; d < DD; d += 32) {
    v8h a_lo = *(const v8h*)(arow + d + a_c0);
    v8h a_hi = *(const v8h*)(arow + d + a_c0 + 8);
    v16h a;
#pragma unroll
    for (int i = 0; i < 8; ++i) { a[i] = a_lo[i]; a[i + 8] = a_hi[i]; }
    v16h b = *(const v16h*)(brow + d + b_k0);
    acc = __builtin_amdgcn_wmma_f32_16x16x32_f16(false, a, false, b,
                                                 (short)0, acc, false, false);
  }

  if (!TRANS) {
    half_t* op = Out + (size_t)(mt * 16 + hb) * N + nt * 16 + l16;
#pragma unroll
    for (int r = 0; r < 8; ++r) op[(size_t)r * N] = (half_t)(acc[r] * oscale);
  } else {
    // Out = Vt[B][N][T]; lane owns column (nt*16+l16), 8 consecutive tokens
    const int grow = mt * 16 + hb;       // global token row of acc[0]
    const int b    = grow / TT;
    const int tl   = grow % TT;
    v8h pk;
#pragma unroll
    for (int r = 0; r < 8; ++r) pk[r] = (half_t)(acc[r] * oscale);
    *(v8h*)(Out + ((size_t)b * N + nt * 16 + l16) * TT + tl) = pk;
  }
}

// ---------------------------------------------------------------------------
// Flash attention. Grid: x = B * (T/64) query tiles, y = 4 column chunks.
// 4 waves/WG; wave owns 16 query rows x 256 output columns.
// Softmax in log2 domain (scale*log2e pre-folded into Q).
// ---------------------------------------------------------------------------
__global__ __launch_bounds__(128) void attn_kernel(
    const half_t* __restrict__ Qh,   // [B*T, HS]  (pre-scaled)
    const half_t* __restrict__ Kh,   // [B*T, HS]
    const half_t* __restrict__ Vt,   // [B][D][T]
    float* __restrict__ Out) {       // [B*T, D]
  __shared__ __align__(32) half_t Pls[4][16][64];  // per-wave P staging

  const int lane = threadIdx.x & 31;
  const int wave = threadIdx.x >> 5;
  const int b    = blockIdx.x >> 6;       // T/64 == 64 query tiles per batch
  const int qt   = blockIdx.x & 63;
  const int colbase = blockIdx.y * 256;

  const int l16  = lane & 15;
  const int a_c0 = (lane >= 16) ? 16 : 0;
  const int b_k0 = (lane >= 16) ? 16 : 0;
  const int hb   = (lane >= 16) ? 8 : 0;

  const int qrow = qt * 64 + wave * 16;   // first query row (within batch)

  // ---- load Q A-frags once (K = 64 -> two 32-chunks) ----
  const half_t* qptr = Qh + ((size_t)b * TT + qrow + l16) * HS;
  v16h qa[2];
#pragma unroll
  for (int c = 0; c < 2; ++c) {
    v8h lo = *(const v8h*)(qptr + c * 32 + a_c0);
    v8h hi = *(const v8h*)(qptr + c * 32 + a_c0 + 8);
#pragma unroll
    for (int i = 0; i < 8; ++i) { qa[c][i] = lo[i]; qa[c][i + 8] = hi[i]; }
  }

  v8f o[16];
#pragma unroll
  for (int i = 0; i < 16; ++i) o[i] = (v8f){};
  float rmax[8], rsum[8];
#pragma unroll
  for (int r = 0; r < 8; ++r) { rmax[r] = -INFINITY; rsum[r] = 0.0f; }

  const half_t* kbase = Kh + (size_t)b * TT * HS;
  const half_t* vbase = Vt + (size_t)b * DD * TT;

  for (int kt = 0; kt <= qt; ++kt) {
    const int kb0 = kt * 64;

    // prefetch next key tile (global_prefetch_b8)
    if (kt < qt)
      __builtin_prefetch(kbase + (size_t)(kb0 + 64 + l16) * HS, 0, 1);

    // ---- S = Q * K^T for 4 subtiles of 16 keys ----
    v8f s[4];
#pragma unroll
    for (int nt = 0; nt < 4; ++nt) {
      const half_t* kp = kbase + (size_t)(kb0 + nt * 16 + l16) * HS;
      v8f acc = {};
#pragma unroll
      for (int c = 0; c < 2; ++c) {
        v16h bfr = *(const v16h*)(kp + c * 32 + b_k0);
        acc = __builtin_amdgcn_wmma_f32_16x16x32_f16(
            false, qa[c], false, bfr, (short)0, acc, false, false);
      }
      s[nt] = acc;
    }

    // ---- causal mask: wave-uniform branch, taken only on diagonal tile ----
    if (kt == qt) {
#pragma unroll
      for (int nt = 0; nt < 4; ++nt) {
        int key = kb0 + nt * 16 + l16;
#pragma unroll
        for (int r = 0; r < 8; ++r)
          if (key > qrow + r + hb) s[nt][r] = -INFINITY;
      }
    }

    // ---- online softmax (log2 domain; rows live across 16-lane halves) ----
    float newmax[8], corr[8];
    bool upd = false;
#pragma unroll
    for (int r = 0; r < 8; ++r) {
      float m = fmaxf(fmaxf(s[0][r], s[1][r]), fmaxf(s[2][r], s[3][r]));
#pragma unroll
      for (int off = 1; off <= 8; off <<= 1)
        m = fmaxf(m, __shfl_xor(m, off, 32));
      newmax[r] = fmaxf(rmax[r], m);
      corr[r]   = __builtin_amdgcn_exp2f(rmax[r] - newmax[r]);
      upd      |= (newmax[r] != rmax[r]);
      rmax[r]   = newmax[r];
    }
#pragma unroll
    for (int r = 0; r < 8; ++r) {
      float acc = 0.0f;
#pragma unroll
      for (int nt = 0; nt < 4; ++nt) {
        float p = __builtin_amdgcn_exp2f(s[nt][r] - newmax[r]);
        s[nt][r] = p;
        acc += p;
      }
#pragma unroll
      for (int off = 1; off <= 8; off <<= 1)
        acc += __shfl_xor(acc, off, 32);
      rsum[r] = rsum[r] * corr[r] + acc;
    }
    // rescale accumulators only when the running max actually moved
    if (__any(upd)) {
#pragma unroll
      for (int i = 0; i < 16; ++i)
#pragma unroll
        for (int r = 0; r < 8; ++r) o[i][r] *= corr[r];
    }

    // ---- P: C-layout -> A-layout via per-wave LDS tile ----
#pragma unroll
    for (int nt = 0; nt < 4; ++nt)
#pragma unroll
      for (int r = 0; r < 8; ++r)
        Pls[wave][r + hb][nt * 16 + l16] = (half_t)s[nt][r];
    asm volatile("s_wait_dscnt 0x0" ::: "memory");

    v16h pa[2];
#pragma unroll
    for (int c = 0; c < 2; ++c) {
      const half_t* pp = &Pls[wave][l16][c * 32 + a_c0];
      v8h lo = *(const v8h*)(pp);
      v8h hi = *(const v8h*)(pp + 8);
#pragma unroll
      for (int i = 0; i < 8; ++i) { pa[c][i] = lo[i]; pa[c][i + 8] = hi[i]; }
    }

    // ---- O += P * V  (16 column tiles x 2 K-chunks = 32 WMMAs) ----
#pragma unroll
    for (int ct = 0; ct < 16; ++ct) {
      const half_t* vp = vbase + (size_t)(colbase + ct * 16 + l16) * TT + kb0;
#pragma unroll
      for (int c = 0; c < 2; ++c) {
        v16h bfr = *(const v16h*)(vp + c * 32 + b_k0);
        o[ct] = __builtin_amdgcn_wmma_f32_16x16x32_f16(
            false, pa[c], false, bfr, (short)0, o[ct], false, false);
      }
    }
  }

  // ---- normalize and store fp32 output ----
  float inv[8];
#pragma unroll
  for (int r = 0; r < 8; ++r) inv[r] = __builtin_amdgcn_rcpf(rsum[r]);
#pragma unroll
  for (int ct = 0; ct < 16; ++ct) {
    float* op = Out + ((size_t)b * TT + qrow + hb) * DD + colbase + ct * 16 + l16;
#pragma unroll
    for (int r = 0; r < 8; ++r) op[(size_t)r * DD] = o[ct][r] * inv[r];
  }
}

// ---------------------------------------------------------------------------
extern "C" void kernel_launch(void* const* d_in, const int* in_sizes, int n_in,
                              void* d_out, int out_size, void* d_ws,
                              size_t ws_size, hipStream_t stream) {
  (void)in_sizes; (void)n_in; (void)out_size; (void)ws_size;
  const float* x  = (const float*)d_in[0];
  const float* Wk = (const float*)d_in[1];
  const float* Wq = (const float*)d_in[2];
  const float* Wv = (const float*)d_in[3];
  float* out = (float*)d_out;

  // workspace layout (bytes)
  char* ws = (char*)d_ws;
  half_t* xh = (half_t*)(ws);                    // B*T*D f16   = 32 MB
  half_t* qh = (half_t*)(ws + (size_t)33554432); // B*T*HS f16  =  2 MB
  half_t* kh = (half_t*)(ws + (size_t)35651584); // B*T*HS f16  =  2 MB
  half_t* vt = (half_t*)(ws + (size_t)37748736); // B*D*T f16   = 32 MB
  half_t* wh = (half_t*)(ws + (size_t)71303168); // weight f16  =  2 MB

  // x -> f16
  {
    int n = BB * TT * DD;
    cvt_f32_to_f16<<<(n / 4 + 255) / 256, 256, 0, stream>>>(x, xh, n);
  }
  // K projection
  {
    int n = HS * DD;
    cvt_f32_to_f16<<<(n / 4 + 255) / 256, 256, 0, stream>>>(Wk, wh, n);
    proj_gemm<0><<<dim3(BB * TT / 16, 1), 128, 0, stream>>>(xh, wh, kh, HS, 1.0f);
  }
  // Q projection (softmax scale * log2e folded in)
  {
    int n = HS * DD;
    cvt_f32_to_f16<<<(n / 4 + 255) / 256, 256, 0, stream>>>(Wq, wh, n);
    proj_gemm<0><<<dim3(BB * TT / 16, 1), 128, 0, stream>>>(xh, wh, qh, HS, QSCALE);
  }
  // V projection, stored transposed as Vt[B][D][T]
  {
    int n = DD * DD;
    cvt_f32_to_f16<<<(n / 4 + 255) / 256, 256, 0, stream>>>(Wv, wh, n);
    proj_gemm<1><<<dim3(BB * TT / 16, DD / 64), 128, 0, stream>>>(xh, wh, vt, DD, 1.0f);
  }
  // causal flash attention
  attn_kernel<<<dim3(BB * (TT / 64), DD / 256), 128, 0, stream>>>(qh, kh, vt, out);
}
